// PairBiasedSelfAttention_66520453481115
// MI455X (gfx1250) — matrix operand order repaired and checked by reading, not compiled, not run on hardware
//
#include <hip/hip_runtime.h>
#include <hip/hip_bf16.h>

// PairBiasedSelfAttention for MI455X (gfx1250, wave32, WMMA + TDM).
// Roofline: streaming z (512MB) dominates => TDM double-buffered bias pass,
// bf16 WMMA everywhere, two-pass bias + flash attention.

#define BB 2
#define LL 1024
#define DMODEL 512
#define NH 8
#define PDIM 64
#define DHEAD 64
#define NEGVAL -10000.0f

typedef __bf16 bf16;
typedef __attribute__((ext_vector_type(16))) __bf16 v16bf;
typedef __attribute__((ext_vector_type(8)))  float  v8f;
typedef __attribute__((ext_vector_type(4)))  unsigned int u32x4;
typedef __attribute__((ext_vector_type(8)))  int i32x8;
typedef __attribute__((ext_vector_type(4)))  int i32x4;

union Frag16 { v16bf v; uint4 q[2]; };

__device__ __forceinline__ int laneid() { return (int)(threadIdx.x & 31); }

__device__ __forceinline__ v8f wmma_bf16(v16bf a, v16bf b, v8f c) {
  return __builtin_amdgcn_wmma_f32_16x16x32_bf16(false, a, false, b, (short)0, c, false, false);
}

// A fragment (16 rows x 32 K) from f32 row-major, converted to bf16.
__device__ __forceinline__ v16bf afrag_f32(const float* __restrict__ base, int ld, int row0, int k0) {
  const int lane = laneid();
  const int m  = lane & 15;
  const int kh = (lane >> 4) << 4;              // 0 or 16
  const float* r = base + (size_t)(row0 + m) * ld + k0;
  v16bf a;
#pragma unroll
  for (int j = 0; j < 8; ++j) a[j]     = (bf16)r[kh + j];
#pragma unroll
  for (int j = 0; j < 8; ++j) a[8 + j] = (bf16)r[kh + 8 + j];
  return a;
}

// A fragment from bf16 row-major source (16B vector loads).
__device__ __forceinline__ v16bf afrag_bf16(const bf16* __restrict__ base, int ld, int row0, int k0) {
  const int lane = laneid();
  const int m  = lane & 15;
  const int kh = (lane >> 4) << 4;
  const bf16* r = base + (size_t)(row0 + m) * ld + k0 + kh;
  Frag16 f;
  f.q[0] = *(const uint4*)(r);
  f.q[1] = *(const uint4*)(r + 8);
  return f.v;
}

// B fragment where source is stored transposed: B[k][n] = M[n0+n][k0+k], K contiguous.
__device__ __forceinline__ v16bf bfrag_bf16_t(const bf16* __restrict__ base, int ld, int n0, int k0) {
  const int lane = laneid();
  const int n  = lane & 15;
  const int kb = (lane >> 4) << 4;              // 0 or 16
  const bf16* p = base + (size_t)(n0 + n) * ld + k0 + kb;
  Frag16 f;
  f.q[0] = *(const uint4*)(p);
  f.q[1] = *(const uint4*)(p + 8);
  return f.v;
}

__device__ __forceinline__ float rowmax16(float v) {
#pragma unroll
  for (int m = 1; m <= 8; m <<= 1) v = fmaxf(v, __shfl_xor(v, m, 32));
  return v;
}
__device__ __forceinline__ float rowsum16(float v) {
#pragma unroll
  for (int m = 1; m <= 8; m <<= 1) v += __shfl_xor(v, m, 32);
  return v;
}

// TDM: 1D tile copy (16 z-rows = 4KB contiguous) global -> LDS.
// D# per CDNA5 ISA: data_size=4B, tile 1024x1.
// 6-arg builtin form (this toolchain): (g0, g1, g2, g3, g4pad, cpol).
__device__ __forceinline__ void tdm_load_4k(unsigned lds_addr, const float* gptr) {
  const unsigned long long ga = (unsigned long long)(const void*)gptr;
  u32x4 g0;
  g0[0] = 1u;                                         // count=1 valid descriptor
  g0[1] = lds_addr;                                   // lds_addr [63:32]
  g0[2] = (unsigned)(ga & 0xffffffffu);               // global_addr[31:0]
  g0[3] = (unsigned)((ga >> 32) & 0x01ffffffu) | (2u << 30);  // addr[56:32] | type=2
  i32x8 g1;
  g1[0] = (2 << 16);                                  // data_size = 4 bytes
  g1[1] = (int)(1024u << 16);                         // tensor_dim0[15:0] = 1024
  g1[2] = (int)(1u << 16);                            // tensor_dim0 hi=0, tensor_dim1 = 1
  g1[3] = (int)(1024u << 16);                         // tensor_dim1 hi=0, tile_dim0 = 1024
  g1[4] = 1;                                          // tile_dim1 = 1, tile_dim2 = 0
  g1[5] = 1024;                                       // tensor_dim0_stride = 1024
  g1[6] = 0;
  g1[7] = 0;
  const i32x4 zero4 = (i32x4)0;
  const i32x8 zero8 = (i32x8)0;
  __builtin_amdgcn_tensor_load_to_lds(g0, g1, zero4, zero4, zero8, 0);
}

// ---------------- Kernel 0: precision/layout prep ----------------
// xbf = bf16(x); Wt[n][k] = bf16(W[k][n]) for Wq,Wk,Wv,Wo.
__global__ void prep_kernel(const float* __restrict__ x,
                            const float* __restrict__ Wq, const float* __restrict__ Wk,
                            const float* __restrict__ Wv, const float* __restrict__ Wo,
                            bf16* __restrict__ xbf, bf16* __restrict__ Wqt, bf16* __restrict__ Wkt,
                            bf16* __restrict__ Wvt, bf16* __restrict__ Wot) {
  const int i = blockIdx.x * blockDim.x + threadIdx.x;     // 1M threads
  xbf[i] = (bf16)x[i];
  if (i < DMODEL * DMODEL) {
    const int nn = i >> 9, kk = i & 511;
    const size_t src = (size_t)kk * DMODEL + nn;
    Wqt[i] = (bf16)Wq[src];
    Wkt[i] = (bf16)Wk[src];
    Wvt[i] = (bf16)Wv[src];
    Wot[i] = (bf16)Wo[src];
  }
}

// ---------------- Kernel 1: QKV projection (16x64 per wave) ----------------
__global__ void qkv_kernel(const bf16* __restrict__ xbf,
                           const bf16* __restrict__ Wqt, const bf16* __restrict__ Wkt,
                           const bf16* __restrict__ Wvt,
                           const float* __restrict__ bq, const float* __restrict__ bk,
                           const float* __restrict__ bv,
                           bf16* __restrict__ Qbf, bf16* __restrict__ Kbf, bf16* __restrict__ Vt) {
  const int w = blockIdx.x * (blockDim.x >> 5) + (threadIdx.x >> 5);
  const int mtile = w & 127;                    // B*L/16 = 128
  const int ng = w >> 7;                        // 0..23 (3 * 512/64)
  const int which = ng >> 3;                    // 0=Q 1=K 2=V
  const int n0 = (ng & 7) << 6;
  const int m0 = mtile << 4;
  const bf16*  Wt = (which == 0) ? Wqt : ((which == 1) ? Wkt : Wvt);
  const float* bi = (which == 0) ? bq  : ((which == 1) ? bk  : bv);

  v8f acc0 = 0.f, acc1 = 0.f, acc2 = 0.f, acc3 = 0.f;
#pragma unroll 2
  for (int k0 = 0; k0 < DMODEL; k0 += 32) {
    const v16bf a = afrag_bf16(xbf, DMODEL, m0, k0);
    acc0 = wmma_bf16(a, bfrag_bf16_t(Wt, DMODEL, n0,      k0), acc0);
    acc1 = wmma_bf16(a, bfrag_bf16_t(Wt, DMODEL, n0 + 16, k0), acc1);
    acc2 = wmma_bf16(a, bfrag_bf16_t(Wt, DMODEL, n0 + 32, k0), acc2);
    acc3 = wmma_bf16(a, bfrag_bf16_t(Wt, DMODEL, n0 + 48, k0), acc3);
  }
  const int lane = laneid();
  const int n = lane & 15, mb = (lane >> 4) << 3;
  v8f accs[4] = {acc0, acc1, acc2, acc3};
#pragma unroll
  for (int t = 0; t < 4; ++t) {
    const int col = n0 + t * 16 + n;
    const int h = col >> 6, d = col & 63;
    const float bval = bi[col];
#pragma unroll
    for (int j = 0; j < 8; ++j) {
      const int row = m0 + mb + j;
      const int b_ = row >> 10, l = row & 1023;
      const float val = accs[t][j] + bval;
      const size_t bh = (size_t)b_ * NH + h;
      if (which == 0)      Qbf[(bh * LL + l) * DHEAD + d] = (bf16)val;
      else if (which == 1) Kbf[(bh * LL + l) * DHEAD + d] = (bf16)val;
      else                 Vt [(bh * DHEAD + d) * LL + l] = (bf16)val;  // V transposed
    }
  }
}

// ---------------- Kernel 2: pair bias = z @ Wpb (TDM double-buffered) ----------------
#define TPW 8   // z tiles (16 rows) per wave
__global__ void bias_kernel(const float* __restrict__ z, const float* __restrict__ Wpb,
                            float* __restrict__ biasbuf) {
  extern __shared__ char smem[];                // 4 waves * 2 bufs * 4KB = 32KB, base off 0
  const int wib = (int)(threadIdx.x >> 5);      // wave in block: 0..3
  const int gw  = blockIdx.x * 4 + wib;
  const int lane = laneid();
  const int n = lane & 15, kb = (lane >> 4) << 4, mb = (lane >> 4) << 3;

  v16bf b0, b1;                                 // Wpb zero-padded 64x8 -> 64x16
#pragma unroll
  for (int j = 0; j < 16; ++j) {
    b0[j] = (n < NH) ? (bf16)Wpb[(kb + j) * NH + n]      : (bf16)0.f;
    b1[j] = (n < NH) ? (bf16)Wpb[(32 + kb + j) * NH + n] : (bf16)0.f;
  }

  const unsigned lds_w = (unsigned)wib * 8192u;
  const float* buf0 = (const float*)(smem + lds_w);
  const float* buf1 = (const float*)(smem + lds_w + 4096);
  const int tile0 = gw * TPW;

  tdm_load_4k(lds_w, z + (size_t)tile0 * 1024);          // prefetch tile 0

  for (int t = 0; t < TPW; ++t) {
    if (t + 1 < TPW)
      tdm_load_4k(lds_w + (((t + 1) & 1) ? 4096u : 0u), z + (size_t)(tile0 + t + 1) * 1024);
    if (t + 1 < TPW) __builtin_amdgcn_s_wait_tensorcnt(1);
    else             __builtin_amdgcn_s_wait_tensorcnt(0);
    asm volatile("" ::: "memory");

    const float* zb = (t & 1) ? buf1 : buf0;    // 16 x 64 f32 tile in LDS
    v8f acc = 0.f;
    acc = wmma_bf16(afrag_f32(zb, PDIM, 0, 0),  b0, acc);
    acc = wmma_bf16(afrag_f32(zb, PDIM, 0, 32), b1, acc);

    if (n < NH) {
      const int r0 = (tile0 + t) << 4;
#pragma unroll
      for (int j = 0; j < 8; ++j) {
        const int r = r0 + mb + j;
        const int b_ = r >> 20;                 // / (L*L)
        const int i  = (r >> 10) & 1023;
        const int jj = r & 1023;
        biasbuf[(((size_t)b_ * NH + n) * LL + i) * LL + jj] = acc[j];
      }
    }
  }
}

// ---------------- Kernel 3: flash attention ----------------
__global__ void attn_kernel(const bf16* __restrict__ Qbf, const bf16* __restrict__ Kbf,
                            const bf16* __restrict__ Vt, const float* __restrict__ biasbuf,
                            const unsigned char* __restrict__ mask, bf16* __restrict__ Obf) {
  __shared__ __align__(16) bf16 slds[16 * 40];  // 16x32 P tile, padded stride
  const int bh = blockIdx.x >> 6;               // b*H + h
  const int qt = blockIdx.x & 63;
  const int b_ = bh >> 3, h = bh & 7;
  const int i0 = qt << 4;
  const bf16*  Qh = Qbf + (size_t)bh * LL * DHEAD;
  const bf16*  Kh = Kbf + (size_t)bh * LL * DHEAD;
  const bf16*  Vh = Vt  + (size_t)bh * DHEAD * LL;
  const float* Bh = biasbuf + (size_t)bh * LL * LL;
  const unsigned char* mk = mask + b_ * LL;

  const int lane = laneid();
  const int n = lane & 15, mb = (lane >> 4) << 3;

  const v16bf q0 = afrag_bf16(Qh, DHEAD, i0, 0);
  const v16bf q1 = afrag_bf16(Qh, DHEAD, i0, 32);

  bool qmasked[8];
#pragma unroll
  for (int j = 0; j < 8; ++j) qmasked[j] = (mk[i0 + mb + j] == 0);

  v8f o0 = 0.f, o1 = 0.f, o2 = 0.f, o3 = 0.f;
  float mrow[8], lrow[8];
#pragma unroll
  for (int j = 0; j < 8; ++j) { mrow[j] = -3.0e38f; lrow[j] = 0.f; }
  const float scale = 0.125f;                   // 1/sqrt(64)

  for (int j0 = 0; j0 < LL; j0 += 32) {
    v8f s0 = 0.f, s1 = 0.f;                     // S = Q K^T (16q x 32k)
    s0 = wmma_bf16(q0, bfrag_bf16_t(Kh, DHEAD, j0,      0),  s0);
    s1 = wmma_bf16(q0, bfrag_bf16_t(Kh, DHEAD, j0 + 16, 0),  s1);
    s0 = wmma_bf16(q1, bfrag_bf16_t(Kh, DHEAD, j0,      32), s0);
    s1 = wmma_bf16(q1, bfrag_bf16_t(Kh, DHEAD, j0 + 16, 32), s1);

    const bool kok0 = (mk[j0 + n] != 0);
    const bool kok1 = (mk[j0 + 16 + n] != 0);
    float v0[8], v1[8];
#pragma unroll
    for (int j = 0; j < 8; ++j) {
      const float* brow = Bh + (size_t)(i0 + mb + j) * LL + j0;
      float t0 = s0[j] * scale + brow[n];
      float t1 = s1[j] * scale + brow[16 + n];
      if (!kok0 || qmasked[j]) t0 = NEGVAL;
      if (!kok1 || qmasked[j]) t1 = NEGVAL;
      v0[j] = t0; v1[j] = t1;
    }

    float p0[8], p1[8];
    v8f av;
#pragma unroll
    for (int j = 0; j < 8; ++j) {
      const float rmax = rowmax16(fmaxf(v0[j], v1[j]));
      const float mnew = fmaxf(mrow[j], rmax);
      const float alpha = __expf(mrow[j] - mnew);
      p0[j] = __expf(v0[j] - mnew);
      p1[j] = __expf(v1[j] - mnew);
      lrow[j] = lrow[j] * alpha + rowsum16(p0[j] + p1[j]);
      mrow[j] = mnew;
      av[j] = alpha;
    }
    o0 *= av; o1 *= av; o2 *= av; o3 *= av;

    // C-layout -> A-layout transpose of P through LDS (single-wave WG: barrier ~free)
    __syncthreads();
#pragma unroll
    for (int j = 0; j < 8; ++j) {
      slds[(mb + j) * 40 + n]      = (bf16)p0[j];
      slds[(mb + j) * 40 + 16 + n] = (bf16)p1[j];
    }
    __syncthreads();
    const v16bf pf = afrag_bf16(slds, 40, 0, 0);

    o0 = wmma_bf16(pf, bfrag_bf16_t(Vh, LL, 0,  j0), o0);
    o1 = wmma_bf16(pf, bfrag_bf16_t(Vh, LL, 16, j0), o1);
    o2 = wmma_bf16(pf, bfrag_bf16_t(Vh, LL, 32, j0), o2);
    o3 = wmma_bf16(pf, bfrag_bf16_t(Vh, LL, 48, j0), o3);
  }

#pragma unroll
  for (int j = 0; j < 8; ++j) {
    const float inv = 1.0f / lrow[j];
    const size_t rowbase = ((size_t)b_ * LL + i0 + mb + j) * DMODEL + h * DHEAD;
    Obf[rowbase + n]      = (bf16)(o0[j] * inv);
    Obf[rowbase + 16 + n] = (bf16)(o1[j] * inv);
    Obf[rowbase + 32 + n] = (bf16)(o2[j] * inv);
    Obf[rowbase + 48 + n] = (bf16)(o3[j] * inv);
  }
}

// ---------------- Kernel 4: output projection (16x64 per wave) ----------------
__global__ void out_kernel(const bf16* __restrict__ Obf, const bf16* __restrict__ Wot,
                           const float* __restrict__ bo, float* __restrict__ out) {
  const int w = blockIdx.x * (blockDim.x >> 5) + (threadIdx.x >> 5);
  const int mtile = w & 127;
  const int ng = w >> 7;                        // 0..7
  const int m0 = mtile << 4, n0 = ng << 6;
  v8f acc0 = 0.f, acc1 = 0.f, acc2 = 0.f, acc3 = 0.f;
#pragma unroll 2
  for (int k0 = 0; k0 < DMODEL; k0 += 32) {
    const v16bf a = afrag_bf16(Obf, DMODEL, m0, k0);
    acc0 = wmma_bf16(a, bfrag_bf16_t(Wot, DMODEL, n0,      k0), acc0);
    acc1 = wmma_bf16(a, bfrag_bf16_t(Wot, DMODEL, n0 + 16, k0), acc1);
    acc2 = wmma_bf16(a, bfrag_bf16_t(Wot, DMODEL, n0 + 32, k0), acc2);
    acc3 = wmma_bf16(a, bfrag_bf16_t(Wot, DMODEL, n0 + 48, k0), acc3);
  }
  const int lane = laneid();
  const int n = lane & 15, mb = (lane >> 4) << 3;
  v8f accs[4] = {acc0, acc1, acc2, acc3};
#pragma unroll
  for (int t = 0; t < 4; ++t) {
    const int col = n0 + t * 16 + n;
    const float bval = bo[col];
#pragma unroll
    for (int j = 0; j < 8; ++j)
      out[(size_t)(m0 + mb + j) * DMODEL + col] = accs[t][j] + bval;
  }
}

extern "C" void kernel_launch(void* const* d_in, const int* in_sizes, int n_in,
                              void* d_out, int out_size, void* d_ws, size_t ws_size,
                              hipStream_t stream) {
  (void)in_sizes; (void)n_in; (void)out_size; (void)ws_size;
  const float* x  = (const float*)d_in[0];
  const float* z  = (const float*)d_in[1];
  const unsigned char* mask = (const unsigned char*)d_in[2];
  const float* Wq = (const float*)d_in[3];
  const float* bq = (const float*)d_in[4];
  const float* Wk = (const float*)d_in[5];
  const float* bk = (const float*)d_in[6];
  const float* Wv = (const float*)d_in[7];
  const float* bv = (const float*)d_in[8];
  const float* Wo = (const float*)d_in[9];
  const float* bo = (const float*)d_in[10];
  const float* Wpb = (const float*)d_in[11];

  char* ws = (char*)d_ws;
  size_t off = 0;
  bf16* xbf = (bf16*)(ws + off); off += (size_t)BB * LL * DMODEL * sizeof(bf16);     // 2 MB
  bf16* Wqt = (bf16*)(ws + off); off += (size_t)DMODEL * DMODEL * sizeof(bf16);      // 0.5 MB
  bf16* Wkt = (bf16*)(ws + off); off += (size_t)DMODEL * DMODEL * sizeof(bf16);
  bf16* Wvt = (bf16*)(ws + off); off += (size_t)DMODEL * DMODEL * sizeof(bf16);
  bf16* Wot = (bf16*)(ws + off); off += (size_t)DMODEL * DMODEL * sizeof(bf16);
  bf16* Qbf = (bf16*)(ws + off); off += (size_t)BB * NH * LL * DHEAD * sizeof(bf16); // 2 MB
  bf16* Kbf = (bf16*)(ws + off); off += (size_t)BB * NH * LL * DHEAD * sizeof(bf16); // 2 MB
  bf16* Vt  = (bf16*)(ws + off); off += (size_t)BB * NH * DHEAD * LL * sizeof(bf16); // 2 MB
  float* biasbuf = (float*)(ws + off); off += (size_t)BB * NH * LL * LL * sizeof(float); // 64 MB
  bf16* Obf = (bf16*)(ws + off); off += (size_t)BB * LL * DMODEL * sizeof(bf16);     // 2 MB

  // prep: 1M threads
  prep_kernel<<<dim3(4096), dim3(256), 0, stream>>>(x, Wq, Wk, Wv, Wo, xbf, Wqt, Wkt, Wvt, Wot);
  // QKV: 128 mtiles * 24 ngroups = 3072 waves; 8 waves/block
  qkv_kernel<<<dim3(384), dim3(256), 0, stream>>>(xbf, Wqt, Wkt, Wvt, bq, bk, bv, Qbf, Kbf, Vt);
  // bias: 131072 tiles / 8 per wave = 16384 waves; 4 waves/block, 32KB dynamic LDS
  bias_kernel<<<dim3(4096), dim3(128), 32768, stream>>>(z, Wpb, biasbuf);
  // attention: one wave per (b,h,qtile) = 1024 single-wave workgroups
  attn_kernel<<<dim3(BB * NH * (LL / 16)), dim3(32), 0, stream>>>(Qbf, Kbf, Vt, biasbuf, mask, Obf);
  // out proj: 128 * 8 = 1024 waves; 8 waves/block
  out_kernel<<<dim3(128), dim3(256), 0, stream>>>(Obf, Wot, bo, (float*)d_out);
}
